// CustomRNN_30142080484115
// MI455X (gfx1250) — compile-verified, matrix operand order checked by reading, not run on hardware
//
#include <hip/hip_runtime.h>
#include <hip/hip_bf16.h>

// ---------------------------------------------------------------------------
// 2-layer LSTM with masking for MI455X (gfx1250, wave32, WMMA).
//   - f16 WMMA (v_wmma_f32_16x16x32_f16) with f32 accumulate.
//   - Wcat = [Wih | Whh] per layer, f16, rows = gate index g (0..4H-1).
//   - Per step, one fused kernel: gates = [x_t | h_{t-1}] * Wcat^T + bias,
//     then hardware v_tanh_f32-based activations + mask blend, lane-locally.
//   - h (f16) double-buffered across steps (all blocks read all of h).
// ---------------------------------------------------------------------------

#define T_DIM 512
#define B_DIM 64
#define IN_DIM 512
#define H_DIM 1024
#define G_DIM (4 * H_DIM)

typedef __attribute__((ext_vector_type(16))) _Float16 v16h;
typedef __attribute__((ext_vector_type(8)))  _Float16 v8h;
typedef __attribute__((ext_vector_type(8)))  float    v8f;

// CDNA5 hardware tanh (TRANS32 op).  The trailing v_nop satisfies the
// "1 independent op after a TRANS op" hazard rule, which the compiler
// cannot schedule for inside inline asm.
__device__ __forceinline__ float tanh_hw(float x) {
    float r;
    asm volatile("v_tanh_f32 %0, %1\n\tv_nop" : "=v"(r) : "v"(x));
    return r;
}

__device__ __forceinline__ float sigmoid_hw(float x) {
    return __builtin_fmaf(0.5f, tanh_hw(0.5f * x), 0.5f);
}

// Load one 16x32 f16 fragment slice for this lane.
// CDNA5 layout (16-bit A/B 16x32): lanes 0-15 hold K = {0..7, 16..23},
// lanes 16-31 hold K = {8..15, 24..31}.  base points at (row, k0);
// koffsel = (lane>>4)*8.
__device__ __forceinline__ v16h load_frag16(const _Float16* __restrict__ base,
                                            int koffsel) {
    v8h lo = *(const v8h*)(base + koffsel);
    v8h hi = *(const v8h*)(base + koffsel + 16);
    return __builtin_shufflevector(lo, hi,
        0, 1, 2, 3, 4, 5, 6, 7, 8, 9, 10, 11, 12, 13, 14, 15);
}

// --------------------------- prep kernels ----------------------------------

__global__ void f32_to_f16_kernel(const float* __restrict__ src,
                                  _Float16* __restrict__ dst, int n) {
    int i = blockIdx.x * blockDim.x + threadIdx.x;
    if (i < n) dst[i] = (_Float16)src[i];
}

// wcat[g][k] = k < D ? Wih[g][k] : Whh[g][k-D]   (f16), n = G*(D+H)
__global__ void build_wcat_kernel(const float* __restrict__ Wih,
                                  const float* __restrict__ Whh,
                                  _Float16* __restrict__ wcat,
                                  int D, int n) {
    int i = blockIdx.x * blockDim.x + threadIdx.x;
    if (i >= n) return;
    int Ktot = D + H_DIM;
    int g = i / Ktot;
    int k = i - g * Ktot;
    float v = (k < D) ? Wih[(size_t)g * D + k]
                      : Whh[(size_t)g * H_DIM + (k - D)];
    wcat[i] = (_Float16)v;
}

__global__ void bias_kernel(const float* __restrict__ bih,
                            const float* __restrict__ bhh,
                            float* __restrict__ bias, int n) {
    int i = blockIdx.x * blockDim.x + threadIdx.x;
    if (i < n) bias[i] = bih[i] + bhh[i];
}

__global__ void zero_state_kernel(float* __restrict__ hbuf,
                                  float* __restrict__ cbuf,
                                  _Float16* __restrict__ hA,
                                  _Float16* __restrict__ hB, int n) {
    int i = blockIdx.x * blockDim.x + threadIdx.x;
    if (i < n) {
        hbuf[i] = 0.0f;
        cbuf[i] = 0.0f;
        hA[i] = (_Float16)0.0f;
        hB[i] = (_Float16)0.0f;
    }
}

__global__ void copy_state_kernel(const float* __restrict__ hbuf,
                                  const float* __restrict__ cbuf,
                                  float* __restrict__ hdst,
                                  float* __restrict__ cdst, int n) {
    int i = blockIdx.x * blockDim.x + threadIdx.x;
    if (i < n) {
        hdst[i] = hbuf[i];
        cdst[i] = cbuf[i];
    }
}

// --------------------------- fused LSTM step -------------------------------
// grid = 32 blocks x 256 threads = 256 waves.
// wave gw: m_tile = gw>>6 (4 tiles of 16 batch rows), n_tile = gw&63
// (64 tiles of 16 hidden cols).  Each wave accumulates four 16x16 gate
// tiles (i, f, g, o) over K = D + H, then does the cell update lane-locally.
__global__ __launch_bounds__(256)
void lstm_step_kernel(const _Float16* __restrict__ x16, int D,
                      const _Float16* __restrict__ wcat,
                      const float* __restrict__ bias,
                      const float* __restrict__ mask,
                      const _Float16* __restrict__ h16_prev,
                      _Float16* __restrict__ h16_next,
                      float* __restrict__ hbuf,
                      float* __restrict__ cbuf,
                      _Float16* __restrict__ next_x,  // layer0: feeds layer1
                      float* __restrict__ out_t,      // layer1: d_out + t*B*H
                      int t) {
    const int lane = threadIdx.x & 31;
    const int wave = threadIdx.x >> 5;
    const int gw = blockIdx.x * 8 + wave;
    const int m_tile = gw >> 6;   // 0..3
    const int n_tile = gw & 63;   // 0..63
    const int Ktot = D + H_DIM;
    const int koffsel = (lane >> 4) << 3;  // 0 or 8
    const int l15 = lane & 15;

    // A source rows: row of [x_t | h_prev] for batch index m_tile*16 + l15
    const _Float16* arow = x16 + ((size_t)t * B_DIM + (m_tile * 16 + l15)) * D;
    const _Float16* hrow = h16_prev + (size_t)(m_tile * 16 + l15) * H_DIM;

    // B source rows: wcat rows for the 4 gates at hidden col ncol
    const int ncol = n_tile * 16 + l15;
    const _Float16* w0 = wcat + (size_t)(0 * H_DIM + ncol) * Ktot;
    const _Float16* w1 = wcat + (size_t)(1 * H_DIM + ncol) * Ktot;
    const _Float16* w2 = wcat + (size_t)(2 * H_DIM + ncol) * Ktot;
    const _Float16* w3 = wcat + (size_t)(3 * H_DIM + ncol) * Ktot;

    v8f acc0 = {}, acc1 = {}, acc2 = {}, acc3 = {};

    // ---- x part of K ----
#pragma unroll 2
    for (int k0 = 0; k0 < D; k0 += 32) {
        v16h a  = load_frag16(arow + k0, koffsel);
        v16h b0 = load_frag16(w0 + k0, koffsel);
        v16h b1 = load_frag16(w1 + k0, koffsel);
        v16h b2 = load_frag16(w2 + k0, koffsel);
        v16h b3 = load_frag16(w3 + k0, koffsel);
        acc0 = __builtin_amdgcn_wmma_f32_16x16x32_f16(
            false, a, false, b0, (short)0, acc0, false, false);
        acc1 = __builtin_amdgcn_wmma_f32_16x16x32_f16(
            false, a, false, b1, (short)0, acc1, false, false);
        acc2 = __builtin_amdgcn_wmma_f32_16x16x32_f16(
            false, a, false, b2, (short)0, acc2, false, false);
        acc3 = __builtin_amdgcn_wmma_f32_16x16x32_f16(
            false, a, false, b3, (short)0, acc3, false, false);
    }
    // ---- h part of K ----
#pragma unroll 2
    for (int k0 = 0; k0 < H_DIM; k0 += 32) {
        v16h a  = load_frag16(hrow + k0, koffsel);
        v16h b0 = load_frag16(w0 + D + k0, koffsel);
        v16h b1 = load_frag16(w1 + D + k0, koffsel);
        v16h b2 = load_frag16(w2 + D + k0, koffsel);
        v16h b3 = load_frag16(w3 + D + k0, koffsel);
        acc0 = __builtin_amdgcn_wmma_f32_16x16x32_f16(
            false, a, false, b0, (short)0, acc0, false, false);
        acc1 = __builtin_amdgcn_wmma_f32_16x16x32_f16(
            false, a, false, b1, (short)0, acc1, false, false);
        acc2 = __builtin_amdgcn_wmma_f32_16x16x32_f16(
            false, a, false, b2, (short)0, acc2, false, false);
        acc3 = __builtin_amdgcn_wmma_f32_16x16x32_f16(
            false, a, false, b3, (short)0, acc3, false, false);
    }

    // C/D layout: VGPR r, lane l -> M = r + 8*(l>>4), N = l&15.
    const float bi = bias[0 * H_DIM + ncol];
    const float bf = bias[1 * H_DIM + ncol];
    const float bg = bias[2 * H_DIM + ncol];
    const float bo = bias[3 * H_DIM + ncol];

#pragma unroll
    for (int r = 0; r < 8; ++r) {
        const int m = m_tile * 16 + r + ((lane >> 4) << 3);  // batch index
        const size_t idx = (size_t)m * H_DIM + ncol;

        float ig = sigmoid_hw(acc0[r] + bi);
        float fg = sigmoid_hw(acc1[r] + bf);
        float gg = tanh_hw(acc2[r] + bg);
        float og = sigmoid_hw(acc3[r] + bo);

        float c_old = cbuf[idx];
        float h_old = hbuf[idx];
        float c_new = fg * c_old + ig * gg;
        float h_new = og * tanh_hw(c_new);

        float mv = mask[t * B_DIM + m];
        h_new = h_new * mv + h_old * (1.0f - mv);
        c_new = c_new * mv + c_old * (1.0f - mv);

        cbuf[idx] = c_new;
        hbuf[idx] = h_new;
        h16_next[idx] = (_Float16)h_new;
        if (next_x) next_x[((size_t)t * B_DIM + m) * H_DIM + ncol] =
            (_Float16)h_new;
        if (out_t) out_t[idx] = h_new;
    }
}

// --------------------------- host side -------------------------------------

extern "C" void kernel_launch(void* const* d_in, const int* in_sizes, int n_in,
                              void* d_out, int out_size, void* d_ws,
                              size_t ws_size, hipStream_t stream) {
    const float* input_ = (const float*)d_in[0];
    const float* mask   = (const float*)d_in[1];
    const float* Wih0   = (const float*)d_in[2];
    const float* Whh0   = (const float*)d_in[3];
    const float* bih0   = (const float*)d_in[4];
    const float* bhh0   = (const float*)d_in[5];
    const float* Wih1   = (const float*)d_in[6];
    const float* Whh1   = (const float*)d_in[7];
    const float* bih1   = (const float*)d_in[8];
    const float* bhh1   = (const float*)d_in[9];
    float* out = (float*)d_out;

    // workspace carve-out
    char* ws = (char*)d_ws;
    size_t off = 0;
    auto alloc = [&](size_t bytes) -> void* {
        off = (off + 255) & ~(size_t)255;
        void* p = ws + off;
        off += bytes;
        return p;
    };
    const int K0 = IN_DIM + H_DIM;   // 1536
    const int K1 = H_DIM + H_DIM;    // 2048
    _Float16* wcat0 = (_Float16*)alloc((size_t)G_DIM * K0 * 2);
    _Float16* wcat1 = (_Float16*)alloc((size_t)G_DIM * K1 * 2);
    float*    bias0 = (float*)alloc((size_t)G_DIM * 4);
    float*    bias1 = (float*)alloc((size_t)G_DIM * 4);
    _Float16* x0    = (_Float16*)alloc((size_t)T_DIM * B_DIM * IN_DIM * 2);
    _Float16* x1    = (_Float16*)alloc((size_t)T_DIM * B_DIM * H_DIM * 2);
    _Float16* h16A  = (_Float16*)alloc((size_t)B_DIM * H_DIM * 2);
    _Float16* h16B  = (_Float16*)alloc((size_t)B_DIM * H_DIM * 2);
    float*    hbuf  = (float*)alloc((size_t)B_DIM * H_DIM * 4);
    float*    cbuf  = (float*)alloc((size_t)B_DIM * H_DIM * 4);

    const int TB = 256;
    auto blocks = [](int n, int tb) { return (n + tb - 1) / tb; };

    // prep
    {
        int n = T_DIM * B_DIM * IN_DIM;
        f32_to_f16_kernel<<<blocks(n, TB), TB, 0, stream>>>(input_, x0, n);
    }
    {
        int n = G_DIM * K0;
        build_wcat_kernel<<<blocks(n, TB), TB, 0, stream>>>(Wih0, Whh0, wcat0,
                                                            IN_DIM, n);
    }
    {
        int n = G_DIM * K1;
        build_wcat_kernel<<<blocks(n, TB), TB, 0, stream>>>(Wih1, Whh1, wcat1,
                                                            H_DIM, n);
    }
    bias_kernel<<<blocks(G_DIM, TB), TB, 0, stream>>>(bih0, bhh0, bias0, G_DIM);
    bias_kernel<<<blocks(G_DIM, TB), TB, 0, stream>>>(bih1, bhh1, bias1, G_DIM);

    const int NS = B_DIM * H_DIM;
    float* hn_out = out + (size_t)T_DIM * B_DIM * H_DIM;
    float* cn_out = hn_out + 2 * (size_t)B_DIM * H_DIM;

    // ---- layer 0 ----
    zero_state_kernel<<<blocks(NS, TB), TB, 0, stream>>>(hbuf, cbuf, h16A,
                                                         h16B, NS);
    for (int t = 0; t < T_DIM; ++t) {
        const _Float16* hprev = (t & 1) ? h16B : h16A;
        _Float16*       hnext = (t & 1) ? h16A : h16B;
        lstm_step_kernel<<<32, 256, 0, stream>>>(
            x0, IN_DIM, wcat0, bias0, mask, hprev, hnext, hbuf, cbuf,
            /*next_x=*/x1, /*out_t=*/(float*)nullptr, t);
    }
    copy_state_kernel<<<blocks(NS, TB), TB, 0, stream>>>(hbuf, cbuf, hn_out,
                                                         cn_out, NS);

    // ---- layer 1 ----
    zero_state_kernel<<<blocks(NS, TB), TB, 0, stream>>>(hbuf, cbuf, h16A,
                                                         h16B, NS);
    for (int t = 0; t < T_DIM; ++t) {
        const _Float16* hprev = (t & 1) ? h16B : h16A;
        _Float16*       hnext = (t & 1) ? h16A : h16B;
        lstm_step_kernel<<<32, 256, 0, stream>>>(
            x1, H_DIM, wcat1, bias1, mask, hprev, hnext, hbuf, cbuf,
            /*next_x=*/(_Float16*)nullptr,
            /*out_t=*/out + (size_t)t * B_DIM * H_DIM, t);
    }
    copy_state_kernel<<<blocks(NS, TB), TB, 0, stream>>>(
        hbuf, cbuf, hn_out + NS, cn_out + NS, NS);
}
